// CoreRNNFW_77489799955120
// MI455X (gfx1250) — matrix operand order repaired and checked by reading, not compiled, not run on hardware
//
#include <hip/hip_runtime.h>
#include <math.h>

// ---------------------------------------------------------------------------
// Fast-weight RNN, factored (linear-attention) formulation for gfx1250.
//   A_t = eta * sum_{s<t} lam^{t-1-s} h_s h_s^T   =>  A q = eta * sum ...
// One workgroup = 16 batch rows for the full T=24 sequence (no global sync).
// GEMMs via v_wmma_f32_16x16x32_bf16 with pre-swizzled bf16 weight fragments.
// ---------------------------------------------------------------------------

typedef __attribute__((ext_vector_type(16))) __bf16 v16bf;
typedef __attribute__((ext_vector_type(8)))  float  v8f;

#define T_STEPS 24
#define BATCH   256
#define DG      256
#define DH      512
#define DOUT    256
#define S_INNER 3
#define LAM     0.95f
#define ETA     0.5f
#define LN_EPS  1e-5f

#define BTILE   16            // batch rows per workgroup (WMMA M)
#define BLOCK   512           // 16 waves of 32
#define NWAVE   16
#define HIST_T  23            // history entries (steps 0..22 update A)

// LDS row strides (elements), chosen so 16 rows hit distinct bank groups
#define LDH     (DH + 8)      // bf16 h rows
#define LDZ     (DG + 8)      // bf16 z rows
#define LDB     (DH + 4)      // f32 h_base rows (also reused for pred 16x256)

// ---------------------------------------------------------------------------
// Weight pre-swizzle: out[((ntile*KC + kc)*32 + lane)*16 + e] =
//   bf16( W[ ntile*16 + (lane&15) ][ kc*32 + (lane>>4)*16 + e ] )
// This is a 32x16 bf16 B-fragment in contiguous K order, matching pack_a (lane = column, K split by lane
// half, 2 K-values per VGPR) so the main kernel loads 32B/lane, coalesced.
// ---------------------------------------------------------------------------
__global__ void pack_kernel(const float* __restrict__ W, __bf16* __restrict__ out,
                            int N, int K) {
  int idx = blockIdx.x * blockDim.x + threadIdx.x;
  if (idx >= N * K) return;
  const int KC    = K >> 5;
  const int e     = idx & 15;
  const int lane  = (idx >> 4) & 31;
  const int rest  = idx >> 9;
  const int kc    = rest % KC;
  const int ntile = rest / KC;
  const int n = ntile * 16 + (lane & 15);
  const int k = kc * 32 + ((lane >> 4) << 4) + e;
  out[idx] = (__bf16)W[(size_t)n * K + k];
}

// A-fragment (16x32 bf16, lane = row, K = 16*(lane>>4) + e, matching B) from LDS
__device__ inline v16bf pack_a(const __bf16* base, int ldr, int kc, int lane) {
  const int m  = lane & 15;
  const int kb = kc * 32 + ((lane >> 4) << 4);
  const __bf16* row = base + m * ldr + kb;
  v16bf a;
#pragma unroll
  for (int e = 0; e < 16; ++e) {
    const int k = e;                       // kb+e
    a[e] = row[k];
  }
  return a;
}

__device__ inline v16bf load_b(const __bf16* frags, int ntile, int kc, int nkc,
                               int lane) {
  const __bf16* p = frags + ((((size_t)ntile * nkc + kc) * 32 + lane) << 4);
  return *(const v16bf*)p;                 // 32B per lane, coalesced
}

__device__ inline float wred(float v) {
#pragma unroll
  for (int m = 16; m >= 1; m >>= 1) v += __shfl_xor(v, m, 32);
  return v;
}

__global__ __launch_bounds__(BLOCK)
void fastweight_kernel(const float* __restrict__ z_seq,
                       const float* __restrict__ clean_vec,
                       const float* __restrict__ b_h,
                       const float* __restrict__ ln_gamma,
                       const float* __restrict__ ln_beta,
                       const float* __restrict__ head_b,
                       const __bf16* __restrict__ whF,
                       const __bf16* __restrict__ wgF,
                       const __bf16* __restrict__ hwF,
                       float* __restrict__ hist,
                       float* __restrict__ partials) {
  __shared__ __bf16 sh_h[BTILE * LDH];     // current h tile (bf16, WMMA A src)
  __shared__ __bf16 sh_z[BTILE * LDZ];     // z_t tile (bf16, WMMA A src)
  __shared__ float  sh_hbase[BTILE * LDB]; // h_base tile (f32); later: pred
  __shared__ float  sh_red[NWAVE];

  const int tid    = threadIdx.x;
  const int wave   = tid >> 5;
  const int lane   = tid & 31;
  const int batch0 = blockIdx.x * BTILE;

  // Per-lane cached LN affine params for columns i = lane + 32*k
  float gam[16], bet[16];
#pragma unroll
  for (int k = 0; k < 16; ++k) {
    gam[k] = ln_gamma[lane + 32 * k];
    bet[k] = ln_beta[lane + 32 * k];
  }

  // h0 = 0
  for (int i = tid; i < BTILE * LDH; i += BLOCK) sh_h[i] = (__bf16)0.f;
  __syncthreads();

  const int r = wave;                                    // this wave's row
  const float* histR = hist + (size_t)(batch0 + r) * HIST_T * DH;

  for (int t = 0; t < T_STEPS; ++t) {
    // ---- load z_t tile into LDS (bf16) ----
    for (int idx = tid; idx < BTILE * DG; idx += BLOCK) {
      const int rr = idx >> 8, cc = idx & (DG - 1);
      sh_z[rr * LDZ + cc] =
          (__bf16)z_seq[((size_t)t * BATCH + batch0 + rr) * DG + cc];
    }
    __syncthreads();

    // ---- h_base = H @ Wh^T + Z @ Wg^T + b_h  (WMMA, 2 n-tiles per wave) ----
#pragma unroll
    for (int half = 0; half < 2; ++half) {
      const int ntile = wave * 2 + half;                 // 0..31 (N = 512)
      v8f acc = {};
#pragma unroll 4
      for (int kc = 0; kc < DH / 32; ++kc) {
        v16bf a = pack_a(sh_h, LDH, kc, lane);
        v16bf b = load_b(whF, ntile, kc, DH / 32, lane);
        acc = __builtin_amdgcn_wmma_f32_16x16x32_bf16(false, a, false, b,
                                                      (short)0, acc, false, false);
      }
#pragma unroll 4
      for (int kc = 0; kc < DG / 32; ++kc) {
        v16bf a = pack_a(sh_z, LDZ, kc, lane);
        v16bf b = load_b(wgF, ntile, kc, DG / 32, lane);
        acc = __builtin_amdgcn_wmma_f32_16x16x32_bf16(false, a, false, b,
                                                      (short)0, acc, false, false);
      }
      const int col = ntile * 16 + (lane & 15);          // C layout: N = lane&15
      const float bh = b_h[col];
#pragma unroll
      for (int v = 0; v < 8; ++v) {                      // M = v + 8*(lane>>4)
        const int m = v + ((lane >> 4) << 3);
        sh_hbase[m * LDB + col] = acc[v] + bh;
      }
    }
    __syncthreads();

    // ---- h = relu(LN(h_base)) : one wave per row ----
    {
      float s1 = 0.f, s2 = 0.f;
#pragma unroll
      for (int k = 0; k < 16; ++k) {
        const float x = sh_hbase[r * LDB + lane + 32 * k];
        s1 += x; s2 += x * x;
      }
      s1 = wred(s1); s2 = wred(s2);
      const float mu   = s1 * (1.f / DH);
      const float rstd = rsqrtf(s2 * (1.f / DH) - mu * mu + LN_EPS);
#pragma unroll
      for (int k = 0; k < 16; ++k) {
        const int i = lane + 32 * k;
        const float y = (sh_hbase[r * LDB + i] - mu) * rstd * gam[k] + bet[k];
        sh_h[r * LDH + i] = (__bf16)fmaxf(y, 0.f);
      }
    }

    // ---- inner loop: h <- relu(LN(h_base + A h)) via rank-1 history ----
    if (t > 0) {
      for (int it = 0; it < S_INNER; ++it) {
        // dots: lane s computes eta*lam^{t-1-s} * (h_s . q)
        float dl = 0.f;
        if (lane < t) {
          const float* hs = histR + (size_t)lane * DH;
          float acc2 = 0.f;
#pragma unroll 8
          for (int i = 0; i < DH; ++i) acc2 += hs[i] * (float)sh_h[r * LDH + i];
          dl = acc2 * ETA * __powf(LAM, (float)(t - 1 - lane));
        }
        // Ah[i] for i = lane + 32*k (coalesced history reads)
        float ah[16];
#pragma unroll
        for (int k = 0; k < 16; ++k) ah[k] = 0.f;
        for (int s = 0; s < t; ++s) {
          const float dv = __shfl(dl, s, 32);
          const float* hs = histR + (size_t)s * DH;
#pragma unroll
          for (int k = 0; k < 16; ++k) ah[k] += dv * hs[lane + 32 * k];
        }
        // LN(h_base + Ah) + relu
        float s1 = 0.f, s2 = 0.f, xr[16];
#pragma unroll
        for (int k = 0; k < 16; ++k) {
          const float x = sh_hbase[r * LDB + lane + 32 * k] + ah[k];
          xr[k] = x; s1 += x; s2 += x * x;
        }
        s1 = wred(s1); s2 = wred(s2);
        const float mu   = s1 * (1.f / DH);
        const float rstd = rsqrtf(s2 * (1.f / DH) - mu * mu + LN_EPS);
#pragma unroll
        for (int k = 0; k < 16; ++k) {
          const int i = lane + 32 * k;
          const float y = (xr[k] - mu) * rstd * gam[k] + bet[k];
          sh_h[r * LDH + i] = (__bf16)fmaxf(y, 0.f);
        }
      }
    }

    // ---- append h_t to history (A update), except final step ----
    if (t < T_STEPS - 1) {
      float* hw = hist + (size_t)(batch0 + r) * HIST_T * DH + (size_t)t * DH;
#pragma unroll
      for (int k = 0; k < 16; ++k) {
        const int i = lane + 32 * k;
        hw[i] = (float)sh_h[r * LDH + i];
      }
    }
    __syncthreads();
  }

  // ---- head: pred = H @ headW^T + head_b (WMMA, 1 n-tile per wave) ----
  {
    const int ntile = wave;                              // N = 256 -> 16 tiles
    v8f acc = {};
#pragma unroll 4
    for (int kc = 0; kc < DH / 32; ++kc) {
      v16bf a = pack_a(sh_h, LDH, kc, lane);
      v16bf b = load_b(hwF, ntile, kc, DH / 32, lane);
      acc = __builtin_amdgcn_wmma_f32_16x16x32_bf16(false, a, false, b,
                                                    (short)0, acc, false, false);
    }
    const int col = ntile * 16 + (lane & 15);
    const float hb = head_b[col];
#pragma unroll
    for (int v = 0; v < 8; ++v) {
      const int m = v + ((lane >> 4) << 3);
      sh_hbase[m * LDB + col] = acc[v] + hb;             // reuse hbase as pred
    }
  }
  __syncthreads();

  // ---- cosine-normalized squared error, one wave per row ----
  {
    const float* pr = sh_hbase + r * LDB;
    const float* cv = clean_vec + (size_t)(batch0 + r) * DOUT;
    float p[8], c[8], sp = 0.f, sc = 0.f;
#pragma unroll
    for (int k = 0; k < 8; ++k) {
      const int i = lane + 32 * k;
      p[k] = pr[i]; c[k] = cv[i];
      sp += p[k] * p[k]; sc += c[k] * c[k];
    }
    sp = wred(sp); sc = wred(sc);
    const float inp = 1.f / (sqrtf(sp) + 1e-6f);
    const float inc = 1.f / (sqrtf(sc) + 1e-6f);
    float e = 0.f;
#pragma unroll
    for (int k = 0; k < 8; ++k) {
      const float d = p[k] * inp - c[k] * inc;
      e += d * d;
    }
    e = wred(e);
    if (lane == 0) sh_red[r] = e;
  }
  __syncthreads();
  if (tid == 0) {
    float s = 0.f;
    for (int w = 0; w < NWAVE; ++w) s += sh_red[w];
    partials[blockIdx.x] = s;
  }
}

__global__ void finalize_kernel(const float* __restrict__ partials,
                                float* __restrict__ out) {
  if (threadIdx.x == 0 && blockIdx.x == 0) {
    float s = 0.f;
    for (int i = 0; i < BATCH / BTILE; ++i) s += partials[i];
    out[0] = s / (float)(BATCH * DOUT);
  }
}

extern "C" void kernel_launch(void* const* d_in, const int* in_sizes, int n_in,
                              void* d_out, int out_size, void* d_ws, size_t ws_size,
                              hipStream_t stream) {
  (void)in_sizes; (void)n_in; (void)out_size; (void)ws_size;
  const float* z_seq  = (const float*)d_in[0];
  const float* clean  = (const float*)d_in[1];
  const float* W_h    = (const float*)d_in[2];
  const float* W_g    = (const float*)d_in[3];
  const float* b_h    = (const float*)d_in[4];
  const float* gamma  = (const float*)d_in[5];
  const float* beta   = (const float*)d_in[6];
  const float* head_W = (const float*)d_in[7];
  const float* head_b = (const float*)d_in[8];

  char* ws = (char*)d_ws;
  __bf16* whF = (__bf16*)(ws);                            // 512 KB
  __bf16* wgF = (__bf16*)(ws + (512 << 10));              // 256 KB
  __bf16* hwF = (__bf16*)(ws + (768 << 10));              // 256 KB
  float* partials = (float*)(ws + (1024 << 10));          // 4 KB slot
  float* hist     = (float*)(ws + (1028 << 10));          // ~11.8 MB

  pack_kernel<<<(DH * DH + 255) / 256, 256, 0, stream>>>(W_h, whF, DH, DH);
  pack_kernel<<<(DH * DG + 255) / 256, 256, 0, stream>>>(W_g, wgF, DH, DG);
  pack_kernel<<<(DOUT * DH + 255) / 256, 256, 0, stream>>>(head_W, hwF, DOUT, DH);

  fastweight_kernel<<<BATCH / BTILE, BLOCK, 0, stream>>>(
      z_seq, clean, b_h, gamma, beta, head_b, whF, wgF, hwF, hist, partials);

  finalize_kernel<<<1, 32, 0, stream>>>(partials, (float*)d_out);
}
